// QnA_25305947308698
// MI455X (gfx1250) — matrix operand order repaired and checked by reading, not compiled, not run on hardware
//
#include <hip/hip_runtime.h>

// ---------------------------------------------------------------------------
// QnA attention block for MI455X (gfx1250, wave32, WMMA).
//   B=16, CIN=D=256, H=W=56 (L=3136), HEADS=8, N=32, KS=3 (K=9), M=4
// Pipeline:
//   0a) fp32 -> bf16 conversion of Wk, Wv, Wproj (row-major, A operand)
//   0b) fused convert+transpose: x (B,256,L) f32 -> xT (B,L,256) bf16
//       so the GEMM B-operand is K-contiguous (two b128 loads per fragment)
//   1) k = Wk @ x   : bf16 WMMA GEMM (reads xT, writes channel-major k)
//   2) v = Wv @ x   : same
//   3) fused QnA attention -> yT (B,L,256) bf16 (contiguous 64B stores/pixel)
//   4) out = Wproj @ y : WMMA GEMM (reads yT), f32 output, reference layout
// GEMM scheme: 4 waves/block; each wave computes a 64x16 output strip
// (4 M-tiles sharing one B fragment per 32-deep k-step). K fully unrolled
// with an explicit double-buffered pipeline: step s+1 fragments are in
// flight while step s's 4 WMMAs execute.
// ---------------------------------------------------------------------------

typedef __bf16 bf16_t;
typedef __attribute__((ext_vector_type(16))) __bf16 v16bf;
typedef __attribute__((ext_vector_type(8)))  __bf16 v8bf;
typedef __attribute__((ext_vector_type(8)))  float  v8f;

#define BATCH 16
#define CIN   256
#define DMODEL 256
#define HEADS 8
#define NDIM  32
#define HH    56
#define WW    56
#define LHW   (HH * WW)          // 3136
#define KSZ   9
#define MTILES 4                 // M-tiles (16 rows each) per wave
#define NWAVES 4                 // waves per block (distinct N-tiles)

static __device__ __forceinline__ bf16_t f2bf(float f) {
  unsigned u = __builtin_bit_cast(unsigned, f);
  unsigned r = u + 0x7FFFu + ((u >> 16) & 1u);       // round-to-nearest-even
  unsigned short h = (unsigned short)(r >> 16);
  return __builtin_bit_cast(bf16_t, h);
}
static __device__ __forceinline__ float bf2f(bf16_t b) {
  unsigned short h = __builtin_bit_cast(unsigned short, b);
  unsigned u = ((unsigned)h) << 16;
  return __builtin_bit_cast(float, u);
}

// -------------------------------- convert ----------------------------------
__global__ void cvt_f32_bf16(const float* __restrict__ src,
                             bf16_t* __restrict__ dst, long n) {
  long i = (long)blockIdx.x * blockDim.x + threadIdx.x;
  long stride = (long)gridDim.x * blockDim.x;
  for (; i < n; i += stride) dst[i] = f2bf(src[i]);
}

// --------------------- fused convert + transpose ---------------------------
// src: (C=256, P=3136) f32 per batch  ->  dst: (P, C) bf16 per batch.
// 32x32 LDS tile, coalesced on both sides. Grid: (P/32=98, C/32=8, BATCH),
// block (32, 8).
__global__ void cvt_transpose(const float* __restrict__ src,
                              bf16_t* __restrict__ dst) {
  __shared__ bf16_t tile[32][33];
  const int p0 = blockIdx.x * 32;
  const int c0 = blockIdx.y * 32;
  const long boff = (long)blockIdx.z * CIN * LHW;
  const int tx = threadIdx.x, ty = threadIdx.y;

  #pragma unroll
  for (int i = 0; i < 4; ++i) {
    const int c = ty + 8 * i;
    tile[c][tx] = f2bf(src[boff + (long)(c0 + c) * LHW + p0 + tx]);
  }
  __syncthreads();
  #pragma unroll
  for (int i = 0; i < 4; ++i) {
    const int p = ty + 8 * i;
    dst[boff + (long)(p0 + p) * CIN + c0 + tx] = tile[tx][p];
  }
}

// ------------------------------ WMMA GEMM ----------------------------------
// C[m, p] = sum_c A[m, c] * XT[p, c]
//   A : 256x256 bf16 row-major (shared across batch)
//   XT: P x 256 bf16 per batch (K-contiguous per spatial position)
//   C : 256 x P (bf16 or f32) per batch
__global__ __launch_bounds__(NWAVES * 32)
void wmma_gemm_256(const bf16_t* __restrict__ A,
                   const bf16_t* __restrict__ XT,
                   void* __restrict__ Cout,
                   int P, int storeF32,
                   long xBatchStride, long cBatchStride) {
  const int lane = threadIdx.x & 31;
  const int wv   = threadIdx.x >> 5;        // wave id within block
  const int n0 = (blockIdx.x * NWAVES + wv) * 16;
  const int m0 = blockIdx.y * (16 * MTILES);
  const int b  = blockIdx.z;

  const int row = lane & 15;                // A row within tile / B column
  const int hi  = lane >> 4;                // 0 or 1
  const int kbA = hi * 8;                   // A k-base per ISA layout
  const int kbB = hi * 8;                   // B k-base per ISA layout
  const int col = row;

  v8f acc[MTILES];
  #pragma unroll
  for (int t = 0; t < MTILES; ++t) acc[t] = (v8f){};

  const bf16_t* __restrict__ Arow = A + (long)(m0 + row) * 256 + kbA;
  const bf16_t* __restrict__ Brow =
      XT + (long)b * xBatchStride + (long)(n0 + col) * 256 + kbB;

  // ---- explicit double-buffered K pipeline (8 steps of 32) ----
  v16bf bcur, bnxt;
  v16bf acur[MTILES], anxt[MTILES];

  {
    v8bf blo = *(const v8bf*)(Brow);
    v8bf bhi = *(const v8bf*)(Brow + 16);
    #pragma unroll
    for (int i = 0; i < 8; ++i) { bcur[i] = blo[i]; bcur[8 + i] = bhi[i]; }
    #pragma unroll
    for (int t = 0; t < MTILES; ++t) {
      v8bf alo = *(const v8bf*)(Arow + t * 16 * 256);        // K kbA..kbA+7
      v8bf ahi = *(const v8bf*)(Arow + t * 16 * 256 + 16);   // K kbA+16..+23
      #pragma unroll
      for (int i = 0; i < 8; ++i) { acur[t][i] = alo[i]; acur[t][8 + i] = ahi[i]; }
    }
  }

  #pragma unroll
  for (int s = 0; s < 8; ++s) {
    const int k0 = s * 32;
    if (s < 7) {
      // Issue next step's fragments before consuming current ones.
      __builtin_prefetch(Brow + k0 + 64, 0, 1);   // global_prefetch_b8
      v8bf blo = *(const v8bf*)(Brow + k0 + 32);
      v8bf bhi = *(const v8bf*)(Brow + k0 + 48);
      #pragma unroll
      for (int i = 0; i < 8; ++i) { bnxt[i] = blo[i]; bnxt[8 + i] = bhi[i]; }
      #pragma unroll
      for (int t = 0; t < MTILES; ++t) {
        const bf16_t* __restrict__ At = Arow + t * 16 * 256 + k0 + 32;
        v8bf alo = *(const v8bf*)(At);
        v8bf ahi = *(const v8bf*)(At + 16);
        #pragma unroll
        for (int i = 0; i < 8; ++i) { anxt[t][i] = alo[i]; anxt[t][8 + i] = ahi[i]; }
      }
    }
    #pragma unroll
    for (int t = 0; t < MTILES; ++t)
      acc[t] = __builtin_amdgcn_wmma_f32_16x16x32_bf16(
          false, acur[t], false, bcur, (short)0, acc[t], false, false);
    if (s < 7) {
      bcur = bnxt;
      #pragma unroll
      for (int t = 0; t < MTILES; ++t) acur[t] = anxt[t];
    }
  }

  // C/D layout: VGPR j -> row (tile_m0 + j) (+8 for lanes 16..31), N = lane&15
  if (storeF32) {
    float* __restrict__ C = (float*)Cout + (long)b * cBatchStride + n0 + col;
    #pragma unroll
    for (int t = 0; t < MTILES; ++t) {
      const int mBase = m0 + t * 16 + hi * 8;
      #pragma unroll
      for (int j = 0; j < 8; ++j)
        C[(long)(mBase + j) * P] = acc[t][j];
    }
  } else {
    bf16_t* __restrict__ C = (bf16_t*)Cout + (long)b * cBatchStride + n0 + col;
    #pragma unroll
    for (int t = 0; t < MTILES; ++t) {
      const int mBase = m0 + t * 16 + hi * 8;
      #pragma unroll
      for (int j = 0; j < 8; ++j)
        C[(long)(mBase + j) * P] = f2bf(acc[t][j]);
    }
  }
}

// --------------------------- fused QnA attention ---------------------------
// One thread per (b, h, l): 9 logits = <q_h, k_window> (zero-padded taps
// contribute 0, matching unfold's zero padding) + pos bias, softmax over 9,
// scale by sum(Wlin). Output written TRANSPOSED: yT[b][l][h*32+n], so each
// thread emits 4 contiguous 16B vector stores.
__global__ void qna_attn(const bf16_t* __restrict__ kten,   // (B,256,L)
                         const bf16_t* __restrict__ vten,   // (B,256,L)
                         const float* __restrict__ q,
                         const float* __restrict__ pos_emb,
                         const float* __restrict__ Wlin,
                         bf16_t* __restrict__ yT) {         // (B,L,256)
  __shared__ float sq[NDIM];
  const int h = blockIdx.y;
  const int b = blockIdx.z;
  if (threadIdx.x < NDIM)
    sq[threadIdx.x] = q[b * DMODEL + h * NDIM + threadIdx.x];
  __syncthreads();

  const int l = blockIdx.x * blockDim.x + threadIdx.x;
  if (l >= LHW) return;
  const int y = l / WW;
  const int x = l % WW;

  const float sW = Wlin[0] + Wlin[1] + Wlin[2] + Wlin[3];
  const int rel[KSZ] = {0, 1, 2, 1, 2, 3, 2, 3, 4};

  const long base = ((long)b * DMODEL + h * NDIM) * LHW;  // k/v channel base

  float attn[KSZ];
  #pragma unroll
  for (int kk = 0; kk < KSZ; ++kk) {
    const int dy = kk / 3 - 1, dx = kk % 3 - 1;
    const int yy = y + dy, xx = x + dx;
    float s = 0.0f;
    if (yy >= 0 && yy < HH && xx >= 0 && xx < WW) {
      const long off = base + yy * WW + xx;
      #pragma unroll
      for (int n = 0; n < NDIM; ++n)
        s += sq[n] * bf2f(kten[off + (long)n * LHW]);
    }
    attn[kk] = s + pos_emb[rel[kk]];
  }

  // softmax over 9, then scale by sum(Wlin)
  float mx = attn[0];
  #pragma unroll
  for (int kk = 1; kk < KSZ; ++kk) mx = fmaxf(mx, attn[kk]);
  float ssum = 0.0f;
  #pragma unroll
  for (int kk = 0; kk < KSZ; ++kk) {
    attn[kk] = __expf(attn[kk] - mx);
    ssum += attn[kk];
  }
  const float inv = sW / ssum;
  #pragma unroll
  for (int kk = 0; kk < KSZ; ++kk) attn[kk] *= inv;

  float o[NDIM];
  #pragma unroll
  for (int n = 0; n < NDIM; ++n) {
    float acc = 0.0f;
    const long nb = base + (long)n * LHW;
    #pragma unroll
    for (int kk = 0; kk < KSZ; ++kk) {
      const int dy = kk / 3 - 1, dx = kk % 3 - 1;
      const int yy = y + dy, xx = x + dx;
      if (yy >= 0 && yy < HH && xx >= 0 && xx < WW)
        acc += attn[kk] * bf2f(vten[nb + yy * WW + xx]);
    }
    o[n] = acc;
  }

  // Packed transposed store: yT[b][l][h*32 .. h*32+31], 4 x 16B.
  bf16_t* __restrict__ dst = yT + ((long)b * LHW + l) * DMODEL + h * NDIM;
  #pragma unroll
  for (int c = 0; c < 4; ++c) {
    v8bf pk;
    #pragma unroll
    for (int i = 0; i < 8; ++i) pk[i] = f2bf(o[c * 8 + i]);
    *(v8bf*)(dst + c * 8) = pk;
  }
}

// ------------------------------- launcher ----------------------------------
extern "C" void kernel_launch(void* const* d_in, const int* in_sizes, int n_in,
                              void* d_out, int out_size, void* d_ws, size_t ws_size,
                              hipStream_t stream) {
  const float* x     = (const float*)d_in[0];  // (16,256,56,56)
  const float* q     = (const float*)d_in[1];  // (16,256,1)
  const float* Wk    = (const float*)d_in[2];  // (256,256)
  const float* Wv    = (const float*)d_in[3];  // (256,256)
  const float* pos   = (const float*)d_in[4];  // (1,5)
  const float* Wlin  = (const float*)d_in[5];  // (1,4)
  const float* Wproj = (const float*)d_in[6];  // (256,256)
  float* out = (float*)d_out;                  // (16,256,56,56) fp32

  const long W_SZ = 256L * 256L;                // 65536
  const long T_SZ = (long)BATCH * DMODEL * LHW; // 12,845,056

  char* ws = (char*)d_ws;
  bf16_t* Wk_bf    = (bf16_t*)(ws);                        // 128 KB each
  bf16_t* Wv_bf    = (bf16_t*)(ws + (1L << 18));
  bf16_t* Wproj_bf = (bf16_t*)(ws + (2L << 18));
  bf16_t* xT_bf    = (bf16_t*)(ws + (3L << 18));           // 25.7 MB each
  bf16_t* k_bf     = xT_bf + T_SZ;                         // (B,256,L)
  bf16_t* v_bf     = k_bf + T_SZ;                          // (B,256,L)
  bf16_t* yT_bf    = v_bf + T_SZ;                          // (B,L,256)

  // 0a) convert weights to bf16 (row-major, A operand)
  cvt_f32_bf16<<<256, 256, 0, stream>>>(Wk, Wk_bf, W_SZ);
  cvt_f32_bf16<<<256, 256, 0, stream>>>(Wv, Wv_bf, W_SZ);
  cvt_f32_bf16<<<256, 256, 0, stream>>>(Wproj, Wproj_bf, W_SZ);

  // 0b) x -> xT bf16 (K-contiguous B operand)
  dim3 tGrid(LHW / 32, CIN / 32, BATCH);       // (98, 8, 16)
  cvt_transpose<<<tGrid, dim3(32, 8), 0, stream>>>(x, xT_bf);

  const long tStride = (long)DMODEL * LHW;     // 256*3136 per batch
  dim3 gemmGrid(LHW / (16 * NWAVES), DMODEL / (16 * MTILES), BATCH); // (49,4,16)

  // 1) k = Wk @ x ;  2) v = Wv @ x   (bf16 channel-major out)
  wmma_gemm_256<<<gemmGrid, NWAVES * 32, 0, stream>>>(
      Wk_bf, xT_bf, k_bf, LHW, 0, tStride, tStride);
  wmma_gemm_256<<<gemmGrid, NWAVES * 32, 0, stream>>>(
      Wv_bf, xT_bf, v_bf, LHW, 0, tStride, tStride);

  // 3) fused attention -> yT (bf16, spatial-major)
  dim3 attnGrid((LHW + 255) / 256, HEADS, BATCH);
  qna_attn<<<attnGrid, 256, 0, stream>>>(k_bf, v_bf, q, pos, Wlin, yT_bf);

  // 4) out = Wproj @ y  (f32 out, reference layout)
  wmma_gemm_256<<<gemmGrid, NWAVES * 32, 0, stream>>>(
      Wproj_bf, yT_bf, out, LHW, 1, tStride, tStride);
}
